// RBFNormalizedNet_48043504173392
// MI455X (gfx1250) — matrix operand order repaired and checked by reading, not compiled, not run on hardware
//
#include <hip/hip_runtime.h>
#include <hip/hip_bf16.h>

typedef __attribute__((ext_vector_type(16))) _Float16 v16h;
typedef __attribute__((ext_vector_type(8)))  _Float16 v8h;
typedef __attribute__((ext_vector_type(8)))  float    v8f;
typedef __attribute__((ext_vector_type(4)))  float    v4f;
typedef __attribute__((ext_vector_type(4)))  unsigned int u32x4;
typedef __attribute__((ext_vector_type(8)))  unsigned int u32x8;

#define N_ROWS  131072
#define C_DIM   1024
#define D_DIM   64
#define OUT_DIM 128
#define MBLK    128   // rows per workgroup (8 waves x 16 rows)
#define CPITCH  72    // centers LDS pitch (f16): 144B rows, 36-bank stride
#define WPITCH  40    // W-slice LDS pitch (f16): 80B rows, 20-bank stride
#define XPITCH  72    // x tile LDS pitch (f16)
#define PPITCH  40    // p staging LDS pitch (f16)

// ---- dynamic LDS carve-up (bytes) ----
#define OFF_C   0          // centers f16 [1024][72]        147456
#define OFF_W   147456     // W K-slices f16 2x[128][40]     20480
#define OFF_X   167936     // x tile f16 [128][72]           18432
#define OFF_P   186368     // p staging f16 8x[16][40]       10240
#define OFF_X2  196608     // x row norms f32 [128]            512
#define OFF_C2  197120     // c2 f32 [1024]                   4096
#define OFF_K   201216     // 1/(2*sigma^2) f32 [1024]        4096
#define SMEM_BYTES 205312

union V16U { v16h v; v8h h[2]; };

static __device__ __forceinline__ v8f wmma_f16(v16h a, v16h b, v8f c) {
    return __builtin_amdgcn_wmma_f32_16x16x32_f16(
        false, a, false, b, (short)0, c, false, false);
}

// Issue one 2D TDM tile load (D# group0 built here, group1 supplied).
// Group2/3 describe a degenerate 3rd dim (tensor_dim2 = 1).
static __device__ __forceinline__ void tdm_load_2d(unsigned int lds_addr,
                                                   unsigned long long gaddr,
                                                   u32x8 g1) {
    u32x4 g0 = { 1u,                                    // count=1, user mode
                 lds_addr,                              // LDS byte address
                 (unsigned int)gaddr,                   // global_addr[31:0]
                 ((unsigned int)(gaddr >> 32) & 0x01FFFFFFu) | 0x80000000u };
                                                        // addr[56:32] | type=2
    u32x4 g2 = { 1u, 0u, 0u, 0u };                      // tensor_dim2 = 1
    u32x4 g3 = { 0u, 0u, 0u, 0u };
    asm volatile("tensor_load_to_lds %0, %1, %2, %3"
                 :: "s"(g0), "s"(g1), "s"(g2), "s"(g3) : "memory");
}

// ---------------------------------------------------------------------------
// Prep: centers -> f16 (row-major == WMMA B feed), c2, k=1/(2 sigma^2), W->f16
// ---------------------------------------------------------------------------
__global__ __launch_bounds__(256) void rbf_prep(
    const float* __restrict__ centers, const float* __restrict__ sigma,
    const float* __restrict__ W,
    _Float16* __restrict__ centersH, _Float16* __restrict__ WH,
    float* __restrict__ c2, float* __restrict__ kArr)
{
    int tid = blockIdx.x * blockDim.x + threadIdx.x;
    if (tid < OUT_DIM * C_DIM) WH[tid] = (_Float16)W[tid];
    if (tid < C_DIM) {
        float s = 0.f;
        #pragma unroll 8
        for (int d = 0; d < D_DIM; ++d) {
            float v = centers[tid * D_DIM + d];
            s += v * v;
            centersH[tid * D_DIM + d] = (_Float16)v;
        }
        c2[tid] = s;
        float sg = sigma[tid];
        kArr[tid] = 1.0f / (2.0f * sg * sg);
    }
}

// ---------------------------------------------------------------------------
// Fused GEMM1 -> softmax (2-pass recompute) -> GEMM2, TDM-staged operands.
// ---------------------------------------------------------------------------
__global__ __launch_bounds__(256) void rbf_main(
    const float* __restrict__ x, const float* __restrict__ bias,
    const _Float16* __restrict__ centersH, const _Float16* __restrict__ WH,
    const float* __restrict__ c2, const float* __restrict__ kArr,
    float* __restrict__ out)
{
    extern __shared__ char smem[];
    _Float16* cLds = (_Float16*)(smem + OFF_C);
    _Float16* wLds = (_Float16*)(smem + OFF_W);
    _Float16* xH   = (_Float16*)(smem + OFF_X);
    _Float16* pst  = (_Float16*)(smem + OFF_P);
    float*    x2s  = (float*)(smem + OFF_X2);
    float*    c2s  = (float*)(smem + OFF_C2);
    float*    ks   = (float*)(smem + OFF_K);
    const unsigned int ldsBase = (unsigned int)(size_t)smem;

    const int tid  = threadIdx.x;
    const int w    = tid >> 5;
    const int lane = tid & 31;
    const int ln   = lane & 15;
    const int hi   = lane >> 4;
    const int m0g  = blockIdx.x * MBLK;
    const int m0   = w * 16;

    // W K-slice descriptor group1: tile 32x128 of a 1024x128 f16 tensor,
    // pad 4 DW every 16 DW  -> 80B LDS pitch.
    const u32x8 g1w = { (1u << 16) | (1u << 20) | (3u << 22) | (3u << 25),
                        (1024u & 0xFFFFu) << 16,   // tensor_dim0 lo16
                        (128u << 16),              // dim0 hi=0 | tensor_dim1 lo16
                        (32u << 16),               // dim1 hi=0 | tile_dim0
                        128u | (1u << 16),         // tile_dim1 | tile_dim2=1
                        1024u, 0u, 0u };           // tensor_dim0_stride

    if (w == 0) {
        // centers: whole 1024x64 f16 tensor as one tile,
        // pad 4 DW every 32 DW -> 144B LDS pitch.
        const u32x8 g1c = { (1u << 16) | (1u << 20) | (4u << 22) | (3u << 25),
                            (64u << 16),
                            (1024u << 16),
                            (64u << 16),
                            1024u | (1u << 16),
                            64u, 0u, 0u };
        tdm_load_2d(ldsBase + OFF_C, (unsigned long long)(size_t)centersH, g1c);
        // first W K-slice into buffer 0
        tdm_load_2d(ldsBase + OFF_W, (unsigned long long)(size_t)WH, g1w);
    }

    // ---- cooperative: x tile f32->f16 + exact row norms; c2/k to LDS ----
    {
        int r = tid >> 1;
        int h = tid & 1;
        const float* src = x + (size_t)(m0g + r) * D_DIM + h * 32;
        float ss = 0.f;
        #pragma unroll
        for (int i = 0; i < 8; ++i) {
            v4f v = *(const v4f*)(src + i * 4);
            ss += v.x * v.x + v.y * v.y + v.z * v.z + v.w * v.w;
            _Float16* dst = &xH[r * XPITCH + h * 32 + i * 4];
            dst[0] = (_Float16)v.x; dst[1] = (_Float16)v.y;
            dst[2] = (_Float16)v.z; dst[3] = (_Float16)v.w;
        }
        ss += __shfl_xor(ss, 1, 32);
        if (h == 0) x2s[r] = ss;
        for (int i = tid; i < C_DIM; i += 256) {
            c2s[i] = c2[i];
            ks[i]  = kArr[i];
        }
    }
    if (w == 0) __builtin_amdgcn_s_wait_tensorcnt(0);  // centers + slice0 in LDS
    __syncthreads();

    // ---- A fragments (rows m0..m0+15, K=0..63), contiguous k = 16*hi + s ----
    V16U a0, a1;
    {
        const _Float16* ab = &xH[(m0 + ln) * XPITCH + hi * 16];
        a0.h[0] = *(const v8h*)(ab);
        a0.h[1] = *(const v8h*)(ab + 8);
        a1.h[0] = *(const v8h*)(ab + 32);
        a1.h[1] = *(const v8h*)(ab + 40);
    }
    float x2r[8];
    #pragma unroll
    for (int r = 0; r < 8; ++r) x2r[r] = x2s[m0 + hi * 8 + r];

    // ================= pass 1: row max over all 1024 centers ================
    float rmax[8];
    #pragma unroll
    for (int r = 0; r < 8; ++r) rmax[r] = -3.0e38f;

    for (int t = 0; t < C_DIM / 16; ++t) {
        int c0 = t * 16;
        const _Float16* cb = &cLds[(c0 + ln) * CPITCH + hi * 16];
        V16U b0, b1;
        b0.h[0] = *(const v8h*)(cb);
        b0.h[1] = *(const v8h*)(cb + 8);
        b1.h[0] = *(const v8h*)(cb + 32);
        b1.h[1] = *(const v8h*)(cb + 40);
        v8f dd = {};
        dd = wmma_f16(a0.v, b0.v, dd);
        dd = wmma_f16(a1.v, b1.v, dd);
        float c2v = c2s[c0 + ln];
        float kv  = ks[c0 + ln];
        #pragma unroll
        for (int r = 0; r < 8; ++r) {
            float lg = (2.f * dd[r] - x2r[r] - c2v) * kv;
            rmax[r] = fmaxf(rmax[r], lg);
        }
    }
    #pragma unroll
    for (int m = 1; m <= 8; m <<= 1)
        #pragma unroll
        for (int r = 0; r < 8; ++r)
            rmax[r] = fmaxf(rmax[r], __shfl_xor(rmax[r], m, 32));

    // ========== pass 2: recompute + exp + fused GEMM2 (lockstep kb) =========
    v8f acc[8] = {};
    float rsum[8];
    #pragma unroll
    for (int r = 0; r < 8; ++r) rsum[r] = 0.f;

    _Float16* myp = &pst[w * 16 * PPITCH];

    for (int kb = 0; kb < C_DIM / 32; ++kb) {
        // launch DMA of next W K-slice into the other buffer (freed by the
        // barrier that ended iteration kb-1); overlaps with compute below
        if (w == 0 && kb + 1 < C_DIM / 32)
            tdm_load_2d(ldsBase + OFF_W + ((kb + 1) & 1) * (MBLK * WPITCH * 2),
                        (unsigned long long)(size_t)WH
                            + (unsigned long long)(kb + 1) * 64ull, g1w);

        #pragma unroll
        for (int sub = 0; sub < 2; ++sub) {
            int c0 = kb * 32 + sub * 16;
            const _Float16* cb = &cLds[(c0 + ln) * CPITCH + hi * 16];
            V16U b0, b1;
            b0.h[0] = *(const v8h*)(cb);
            b0.h[1] = *(const v8h*)(cb + 8);
            b1.h[0] = *(const v8h*)(cb + 32);
            b1.h[1] = *(const v8h*)(cb + 40);
            v8f dd = {};
            dd = wmma_f16(a0.v, b0.v, dd);
            dd = wmma_f16(a1.v, b1.v, dd);
            float c2v = c2s[c0 + ln];
            float kv  = ks[c0 + ln];
            #pragma unroll
            for (int r = 0; r < 8; ++r) {
                float lg = (2.f * dd[r] - x2r[r] - c2v) * kv;
                float p  = __expf(lg - rmax[r]);
                rsum[r] += p;
                myp[(r + hi * 8) * PPITCH + sub * 16 + ln] = (_Float16)p;
            }
        }
        asm volatile("s_wait_dscnt 0" ::: "memory");  // p stage visible to wave

        V16U a2;
        const _Float16* pb = &myp[ln * PPITCH + hi * 16];
        a2.h[0] = *(const v8h*)(pb);
        a2.h[1] = *(const v8h*)(pb + 8);

        const _Float16* wb = &wLds[(kb & 1) * (MBLK * WPITCH) + hi * 16];
        #pragma unroll
        for (int j = 0; j < 8; ++j) {
            const _Float16* wr = wb + (j * 16 + ln) * WPITCH;
            V16U bw;
            bw.h[0] = *(const v8h*)(wr);
            bw.h[1] = *(const v8h*)(wr + 8);
            acc[j] = wmma_f16(a2.v, bw.v, acc[j]);
        }

        if (w == 0) __builtin_amdgcn_s_wait_tensorcnt(0);  // next slice landed
        __syncthreads();  // all waves done with wLds[kb&1]; slice kb+1 visible
    }

    // ---- row sums, normalize, bias, store ----
    #pragma unroll
    for (int m = 1; m <= 8; m <<= 1)
        #pragma unroll
        for (int r = 0; r < 8; ++r)
            rsum[r] += __shfl_xor(rsum[r], m, 32);
    float rinv[8];
    #pragma unroll
    for (int r = 0; r < 8; ++r) rinv[r] = 1.0f / rsum[r];

    #pragma unroll
    for (int j = 0; j < 8; ++j) {
        float bv = bias[j * 16 + ln];
        #pragma unroll
        for (int r = 0; r < 8; ++r) {
            int row = m0g + m0 + hi * 8 + r;
            out[(size_t)row * OUT_DIM + j * 16 + ln] = acc[j][r] * rinv[r] + bv;
        }
    }
}

// ---------------------------------------------------------------------------
extern "C" void kernel_launch(void* const* d_in, const int* in_sizes, int n_in,
                              void* d_out, int out_size, void* d_ws, size_t ws_size,
                              hipStream_t stream) {
    const float* x       = (const float*)d_in[0];
    const float* centers = (const float*)d_in[1];
    const float* sigma   = (const float*)d_in[2];
    const float* W       = (const float*)d_in[3];
    const float* bias    = (const float*)d_in[4];
    float* out = (float*)d_out;

    char* ws = (char*)d_ws;
    _Float16* centersH = (_Float16*)(ws);                       // 128 KB
    _Float16* WH       = (_Float16*)(ws + 131072);              // 256 KB
    float*    c2       = (float*)(ws + 131072 + 262144);        //   4 KB
    float*    kArr     = (float*)(ws + 131072 + 262144 + 4096); //   4 KB

    rbf_prep<<<(OUT_DIM * C_DIM) / 256, 256, 0, stream>>>(
        centers, sigma, W, centersH, WH, c2, kArr);
    rbf_main<<<N_ROWS / MBLK, 256, SMEM_BYTES, stream>>>(
        x, bias, centersH, WH, c2, kArr, out);
}
